// SRNets_80032420594220
// MI455X (gfx1250) — matrix operand order repaired and checked by reading, not compiled, not run on hardware
//
#include <hip/hip_runtime.h>
#include <hip/hip_bf16.h>

// ---------------------------------------------------------------------------
// MuLUT SRNet, MI455X (gfx1250, wave32).
// Compute-bound (~225 GFLOP vs ~35 MB traffic) -> all 64x64 1x1 convs run on
// v_wmma_f32_16x16x32_f16 with activations/weights staged in LDS (f16 data,
// f32 accumulate to survive the round(out*127) quantization).
// Rotation loop is kept rolled (r is wave-uniform) to stay under 256 VGPRs
// and avoid s_set_vgpr_msb churn / occupancy loss.
// ---------------------------------------------------------------------------

typedef __attribute__((ext_vector_type(16))) _Float16 v16h;
typedef __attribute__((ext_vector_type(8)))  _Float16 v8h;
typedef __attribute__((ext_vector_type(8)))  float    v8f;

#define NFC   64
#define HH_   256
#define WW_   256
#define BB_   8
#define WAVES 4
#define PXW   16   // pixels per wave

// tap offsets for rotation r, tap k = kh*2+kw (conv weight index [kh][kw]):
// r0:(kh,kw)  r1:(kw,-kh)  r2:(-kh,-kw)  r3:(-kw,kh), edge-clamped.
__device__ static constexpr int DH_[4][4] = {
  {0,0, 1, 1}, {0,1, 0, 1}, {0,0,-1,-1}, {0,-1,0,-1}};
__device__ static constexpr int DW_[4][4] = {
  {0,1, 0, 1}, {0,0,-1,-1}, {0,-1,0,-1}, {0,0, 1, 1}};

// pixel-shuffle subpixel (su,sv)=c/4,c%4 rotated back to original frame
// (r is wave-uniform -> uniform branch)
__device__ inline int remap_oc(int r, int c) {
  int su = c >> 2, sv = c & 3;
  switch (r) {
    case 0:  return (su << 2) | sv;
    case 1:  return (sv << 2) | (3 - su);
    case 2:  return ((3 - su) << 2) | (3 - sv);
    default: return ((3 - sv) << 2) | su;
  }
}

// A tile (16x32 f16): lanes0-15 = rows M (k 0..7 then 16..23), lanes16-31 k+8.
// W stored [out][in] f16 in LDS -> two b128 loads.
__device__ inline v16h load_a_tile(const _Float16* Wl, int lane, int m, int slice) {
  int row = (m << 4) + (lane & 15);
  int k0  = (slice << 5) + ((lane & 16) ? 8 : 0);
  const _Float16* p = Wl + row * NFC + k0;
  v8h lo = *reinterpret_cast<const v8h*>(p);
  v8h hi = *reinterpret_cast<const v8h*>(p + 16);
  v16h a;
#pragma unroll
  for (int i = 0; i < 8; ++i) { a[i] = lo[i]; a[i + 8] = hi[i]; }
  return a;
}

// B tile (32x16 f16): lane = column (pixel); lanes0-15 hold k 0..7 then 16..23
// of the slice, lanes16-31 k+8 (matches A tile, act[px][ch]) -> two b128.
__device__ inline v16h load_b_tile(const _Float16* act, int lane, int slice) {
  int px = lane & 15;
  int k0 = (slice << 5) + ((lane & 16) ? 8 : 0);
  const _Float16* p = act + px * NFC + k0;
  v8h lo = *reinterpret_cast<const v8h*>(p);
  v8h hi = *reinterpret_cast<const v8h*>(p + 16);
  v16h b;
#pragma unroll
  for (int i = 0; i < 8; ++i) { b[i] = lo[i]; b[i + 8] = hi[i]; }
  return b;
}

// one 64->64 1x1 conv layer for 16 pixels (one wave): 8 WMMAs, bias+ReLU,
// f16 repack in place.  D layout: lane = pixel column, VGPR q = out channel.
__device__ inline void layer_wmma64(_Float16* actw, const _Float16* Wl,
                                    const float* bias, int lane) {
  v16h b0 = load_b_tile(actw, lane, 0);   // all B read before any store
  v16h b1 = load_b_tile(actw, lane, 1);
  const int px  = lane & 15;
  const int chb = (lane & 16) ? 8 : 0;
#pragma unroll
  for (int m = 0; m < 4; ++m) {
    v8f c = {};
    v16h a0 = load_a_tile(Wl, lane, m, 0);
    c = __builtin_amdgcn_wmma_f32_16x16x32_f16(false, a0, false, b0,
                                               (short)0, c, false, false);
    v16h a1 = load_a_tile(Wl, lane, m, 1);
    c = __builtin_amdgcn_wmma_f32_16x16x32_f16(false, a1, false, b1,
                                               (short)0, c, false, false);
    v8h o;
#pragma unroll
    for (int q = 0; q < 8; ++q) {
      float v = c[q] + bias[(m << 4) + chb + q];
      o[q] = (_Float16)fmaxf(v, 0.f);
    }
    *reinterpret_cast<v8h*>(actw + px * NFC + (m << 4) + chb) = o;
  }
}

template <int STAGE>
__global__ void __launch_bounds__(128)
srnet_stage(const float* __restrict__ img,
            const float* __restrict__ w1g, const float* __restrict__ b1g,
            const float* __restrict__ w2g, const float* __restrict__ b2g,
            const float* __restrict__ w3g, const float* __restrict__ b3g,
            const float* __restrict__ w4g, const float* __restrict__ b4g,
            float* __restrict__ out) {
  __shared__ __attribute__((aligned(16))) _Float16 w2h[NFC * NFC];
  __shared__ __attribute__((aligned(16))) _Float16 w3h[NFC * NFC];
  __shared__ __attribute__((aligned(16))) _Float16 w4h[STAGE == 0 ? 8 : 16 * NFC];
  __shared__ float w1f[NFC * 4];
  __shared__ float b1f[NFC], b2f[NFC], b3f[NFC];
  __shared__ float w4f[STAGE == 0 ? NFC : 1];
  __shared__ float b4f[16];
  __shared__ __attribute__((aligned(16))) _Float16 act[WAVES][PXW * NFC];
  __shared__ float acc[WAVES][STAGE == 0 ? PXW : PXW * 16];

  const int tid  = threadIdx.x;
  const int wave = tid >> 5;
  const int lane = tid & 31;
  const int px   = lane & 15;
  const int g    = blockIdx.x * (WAVES * PXW) + wave * PXW + px;
  const int bb   = g >> 16;
  const int h    = (g >> 8) & 255;
  const int w    = g & 255;

  float* accw = acc[wave];
  const int ACCN = (STAGE == 0 ? PXW : PXW * 16);
  for (int i = lane; i < ACCN; i += 32) accw[i] = 0.f;

  _Float16* actw = act[wave];

  for (int s = 0; s < 3; ++s) {                 // samplers
    __syncthreads();
    for (int i = tid; i < NFC * 4; i += 128) w1f[i] = w1g[s * NFC * 4 + i];
    for (int i = tid; i < NFC; i += 128) {
      b1f[i] = b1g[s * NFC + i];
      b2f[i] = b2g[s * NFC + i];
      b3f[i] = b3g[s * NFC + i];
    }
    for (int i = tid; i < NFC * NFC; i += 128) {
      w2h[i] = (_Float16)w2g[s * NFC * NFC + i];
      w3h[i] = (_Float16)w3g[s * NFC * NFC + i];
    }
    if (STAGE == 0) {
      for (int i = tid; i < NFC; i += 128) w4f[i] = w4g[s * NFC + i];
      if (tid == 0) b4f[0] = b4g[s];
    } else {
      for (int i = tid; i < 16 * NFC; i += 128)
        w4h[i] = (_Float16)w4g[s * 16 * NFC + i];
      if (tid < 16) b4f[tid] = b4g[s * 16 + tid];
    }
    __syncthreads();

#pragma unroll 1
    for (int r = 0; r < 4; ++r) {               // rotations (kept rolled)
      // ---- layer 1: 2x2 conv (1->64), VALU, edge-clamped rotated taps ----
      float t[4];
#pragma unroll
      for (int k = 0; k < 4; ++k) {
        int hh = h + DH_[r][k]; hh = hh < 0 ? 0 : (hh > HH_ - 1 ? HH_ - 1 : hh);
        int ww = w + DW_[r][k]; ww = ww < 0 ? 0 : (ww > WW_ - 1 ? WW_ - 1 : ww);
        t[k] = img[(bb << 16) + (hh << 8) + ww];
      }
      const int c0 = (lane & 16) ? 32 : 0;      // lane pair splits 64 channels
#pragma unroll
      for (int cc = 0; cc < 32; cc += 8) {
        v8h o;
#pragma unroll
        for (int j = 0; j < 8; ++j) {
          int c = c0 + cc + j;
          float v = b1f[c];
#pragma unroll
          for (int k = 0; k < 4; ++k) v += w1f[c * 4 + k] * t[k];
          o[j] = (_Float16)fmaxf(v, 0.f);
        }
        *reinterpret_cast<v8h*>(actw + px * NFC + c0 + cc) = o;
      }

      // ---- layers 2,3: 64x64 1x1 convs on WMMA ----
      layer_wmma64(actw, w2h, b2f, lane);
      layer_wmma64(actw, w3h, b3f, lane);

      // ---- layer 4 ----
      if (STAGE == 0) {                          // 64 -> 1: VALU dot
        const int half = (lane >> 4) & 1;
        float sum = 0.f;
        const _Float16* a = actw + px * NFC + half * 32;
#pragma unroll
        for (int i = 0; i < 32; ++i) sum += (float)a[i] * w4f[half * 32 + i];
        sum += __shfl_xor(sum, 16, 32);
        float val = rintf((sum + b4f[0]) * 127.f);
        if (lane < 16) accw[px] += val;
      } else {                                   // 64 -> 16: 2 WMMAs
        v16h bt0 = load_b_tile(actw, lane, 0);
        v16h bt1 = load_b_tile(actw, lane, 1);
        v8f c = {};
        v16h a0 = load_a_tile(w4h, lane, 0, 0);
        c = __builtin_amdgcn_wmma_f32_16x16x32_f16(false, a0, false, bt0,
                                                   (short)0, c, false, false);
        v16h a1 = load_a_tile(w4h, lane, 0, 1);
        c = __builtin_amdgcn_wmma_f32_16x16x32_f16(false, a1, false, bt1,
                                                   (short)0, c, false, false);
        const int chb = (lane & 16) ? 8 : 0;
#pragma unroll
        for (int q = 0; q < 8; ++q) {
          int ch = chb + q;
          float val = rintf((c[q] + b4f[ch]) * 127.f);
          accw[px * 16 + remap_oc(r, ch)] += val;   // rotated pixel-shuffle
        }
      }
    }
  }

  // ---- finalize ----
  if (STAGE == 0) {
    if (lane < 16) {
      float v = accw[px] * (1.f / 12.f) + 127.f;   // / (NUM_SAMPLERS*4)
      v = fminf(fmaxf(v, 0.f), 255.f);
      float x1 = rintf(v) * (1.f / 255.f);
      out[g] = x1 + img[g];                        // residual sum for stage 1
    }
  } else {
    const int chb = (lane & 16) ? 8 : 0;
#pragma unroll
    for (int q = 0; q < 8; ++q) {
      int oc = chb + q;
      float v = rintf(accw[px * 16 + oc] * (1.f / 3.f)) * (1.f / 255.f);
      int oh = (h << 2) + (oc >> 2);
      int ow = (w << 2) + (oc & 3);
      out[((size_t)bb << 20) + ((size_t)oh << 10) + ow] = v;
    }
  }
}

extern "C" void kernel_launch(void* const* d_in, const int* in_sizes, int n_in,
                              void* d_out, int out_size, void* d_ws, size_t ws_size,
                              hipStream_t stream) {
  (void)in_sizes; (void)n_in; (void)out_size; (void)ws_size;
  const float* x = (const float*)d_in[0];
  const float* s0[8];
  const float* s1[8];
  for (int i = 0; i < 8; ++i) {
    s0[i] = (const float*)d_in[1 + i];
    s1[i] = (const float*)d_in[9 + i];
  }
  float* resid = (float*)d_ws;  // 8*256*256 floats: x1 + x_orig

  dim3 grid(BB_ * HH_ * WW_ / (WAVES * PXW));
  dim3 block(128);
  srnet_stage<0><<<grid, block, 0, stream>>>(
      x, s0[0], s0[1], s0[2], s0[3], s0[4], s0[5], s0[6], s0[7], resid);
  srnet_stage<1><<<grid, block, 0, stream>>>(
      resid, s1[0], s1[1], s1[2], s1[3], s1[4], s1[5], s1[6], s1[7],
      (float*)d_out);
}
